// Net_42004780155007
// MI455X (gfx1250) — compile-verified, matrix operand order checked by reading, not run on hardware
//
#include <hip/hip_runtime.h>
#include <math.h>

// ---------------------------------------------------------------------------
// Types for CDNA5 WMMA
// ---------------------------------------------------------------------------
typedef __bf16 v16bf __attribute__((ext_vector_type(16)));
typedef float  v8f   __attribute__((ext_vector_type(8)));
typedef int    v4i   __attribute__((ext_vector_type(4)));

#define BATCH 8192
#define DFEAT 4611
#define KPAD  4640      // 145 * 32
#define KSTEPS 145

// Async global->LDS support (gfx1250): guarded so compilation can't fail if
// the toolchain doesn't declare the builtins; falls back to sync staging.
#if defined(__has_builtin)
#  if __has_builtin(__builtin_amdgcn_global_load_async_to_lds_b128)
#    define HAVE_ASYNC_LDS 1
#  else
#    define HAVE_ASYNC_LDS 0
#  endif
#  if __has_builtin(__builtin_amdgcn_s_wait_asynccnt)
#    define HAVE_WAIT_ASYNC 1
#  else
#    define HAVE_WAIT_ASYNC 0
#  endif
#else
#  define HAVE_ASYNC_LDS 0
#  define HAVE_WAIT_ASYNC 0
#endif

#if HAVE_ASYNC_LDS
typedef __attribute__((address_space(1))) v4i* as1_v4i_ptr;
typedef __attribute__((address_space(3))) v4i* as3_v4i_ptr;
typedef const __attribute__((address_space(1))) void* as1_cvoid_ptr;
typedef __attribute__((address_space(3))) void* as3_void_ptr;
#endif

// ---------------------------------------------------------------------------
// Small helpers
// ---------------------------------------------------------------------------
__device__ __forceinline__ unsigned short f2bf(float f) {
    // round-to-nearest-even fp32 -> bf16 (finite inputs)
    unsigned u = __float_as_uint(f);
    u += 0x7FFFu + ((u >> 16) & 1u);
    return (unsigned short)(u >> 16);
}

__device__ __forceinline__ unsigned f2ord(float f) {
    // order-preserving float -> uint key (for atomicMin)
    unsigned b = __float_as_uint(f);
    return (b & 0x80000000u) ? ~b : (b | 0x80000000u);
}
__device__ __forceinline__ float ord2f(unsigned k) {
    unsigned b = (k & 0x80000000u) ? (k & 0x7FFFFFFFu) : ~k;
    return __uint_as_float(b);
}

__device__ __forceinline__ float eluf(float x) {
    return x > 0.f ? x : (expf(x) - 1.f);
}

__device__ __forceinline__ float wave_reduce_sum(float v) {
    #pragma unroll
    for (int o = 16; o > 0; o >>= 1) v += __shfl_xor(v, o);
    return v;
}

// 16-byte per-lane copy into LDS: async DMA when available, else sync.
__device__ __forceinline__ void cp16_to_lds(const unsigned short* gsrc,
                                            unsigned short* ldst) {
#if HAVE_ASYNC_LDS
    as1_cvoid_ptr g1 = (as1_cvoid_ptr)gsrc;       // generic -> AS(1)
    as3_void_ptr  l3 = (as3_void_ptr)ldst;        // generic -> AS(3)
    __builtin_amdgcn_global_load_async_to_lds_b128(
        (as1_v4i_ptr)g1, (as3_v4i_ptr)l3, 0, 0);
#else
    *(uint4*)ldst = *(const uint4*)gsrc;
#endif
}

__device__ __forceinline__ void wait_async_le(int n) {
#if HAVE_ASYNC_LDS
#  if HAVE_WAIT_ASYNC
    if (n) __builtin_amdgcn_s_wait_asynccnt(4);
    else   __builtin_amdgcn_s_wait_asynccnt(0);
#  else
    if (n) asm volatile("s_wait_asynccnt 0x4" ::: "memory");
    else   asm volatile("s_wait_asynccnt 0x0" ::: "memory");
#  endif
#else
    (void)n;
#endif
}

// Generate one element of G = concat([W1,B1,W2,B2,W3,B3]) from noise row ns[25]
__device__ __forceinline__ float gen_g(int d, const float* ns,
    const float* w1W, const float* w1b, const float* b1W, const float* b1b,
    const float* w2W, const float* w2b, const float* b2W, const float* b2b,
    const float* w3W, const float* w3b, const float* b3W, const float* b3b)
{
    const float* W; const float* bias; int off, C;
    if (d < 192)       { W = w1W; bias = w1b; off = d;        C = 192;  }
    else if (d < 256)  { W = b1W; bias = b1b; off = d - 192;  C = 64;   }
    else if (d < 4352) { W = w2W; bias = w2b; off = d - 256;  C = 4096; }
    else if (d < 4416) { W = b2W; bias = b2b; off = d - 4352; C = 64;   }
    else if (d < 4608) { W = w3W; bias = w3b; off = d - 4416; C = 192;  }
    else               { W = b3W; bias = b3b; off = d - 4608; C = 3;    }
    float acc = bias[off];
    #pragma unroll
    for (int k = 0; k < 25; ++k) acc += ns[k] * W[k * C + off];
    return acc;
}

// ---------------------------------------------------------------------------
// Init: zero norms/sums, saturate min keys, seed kl with log(B/(B-1))
// ---------------------------------------------------------------------------
__global__ void init_kernel(float* gn, float* pn, unsigned* wpm, unsigned* wwm,
                            float* out) {
    int i = blockIdx.x * 256 + threadIdx.x;
    if (i < BATCH) {
        gn[i] = 0.f; pn[i] = 0.f;
        wpm[i] = 0xFFFFFFFFu; wwm[i] = 0xFFFFFFFFu;
    }
    if (i == 0) {
        out[0] = 0.f;
        out[1] = logf((float)BATCH / (float)(BATCH - 1));
    }
}

// ---------------------------------------------------------------------------
// Hypernet: G[b,d] (fp32, exact) -> bf16 padded buffer, accumulate ||G_b||^2
// ---------------------------------------------------------------------------
__global__ __launch_bounds__(256) void hyper_kernel(
    const float* __restrict__ nsrc,
    const float* w1W, const float* w1b, const float* b1W, const float* b1b,
    const float* w2W, const float* w2b, const float* b2W, const float* b2b,
    const float* w3W, const float* w3b, const float* b3W, const float* b3b,
    unsigned short* __restrict__ Gb, float* __restrict__ gn)
{
    int b = blockIdx.y;
    int d = blockIdx.x * 256 + threadIdx.x;
    __shared__ float ns[25];
    if (threadIdx.x < 25) ns[threadIdx.x] = nsrc[b * 25 + threadIdx.x];
    __syncthreads();
    float sq = 0.f;
    if (d < KPAD) {
        float v = 0.f;
        if (d < DFEAT) {
            v = gen_g(d, ns, w1W, w1b, b1W, b1b, w2W, w2b, b2W, b2b,
                      w3W, w3b, b3W, b3b);
            sq = v * v;
        }
        Gb[(size_t)b * KPAD + d] = f2bf(v);
    }
    float s = wave_reduce_sum(sq);
    if ((threadIdx.x & 31) == 0) atomicAdd(&gn[b], s);
}

// ---------------------------------------------------------------------------
// Prior: fp32 -> bf16 padded buffer, accumulate ||prior_i||^2
// ---------------------------------------------------------------------------
__global__ __launch_bounds__(256) void prior_kernel(
    const float* __restrict__ prior, unsigned short* __restrict__ Pb,
    float* __restrict__ pn)
{
    int b = blockIdx.y;
    int d = blockIdx.x * 256 + threadIdx.x;
    float sq = 0.f;
    if (d < KPAD) {
        float v = (d < DFEAT) ? prior[(size_t)b * DFEAT + d] : 0.f;
        Pb[(size_t)b * KPAD + d] = f2bf(v);
        sq = v * v;
    }
    float s = wave_reduce_sum(sq);
    if ((threadIdx.x & 31) == 0) atomicAdd(&pn[b], s);
}

// ---------------------------------------------------------------------------
// Decode: per-sample bmm chain, recomputed exactly in fp32 from n.
// Writes z_hat to out[2..] and accumulates recon into out[0].
// ---------------------------------------------------------------------------
__global__ __launch_bounds__(64) void decode_kernel(
    const float* __restrict__ z, const float* __restrict__ z_next,
    const float* __restrict__ nsrc,
    const float* w1W, const float* w1b, const float* b1W, const float* b1b,
    const float* w2W, const float* w2b, const float* b2W, const float* b2b,
    const float* w3W, const float* w3b, const float* b3W, const float* b3b,
    float* __restrict__ out)
{
    int b = blockIdx.x, t = threadIdx.x;
    __shared__ float ns[25], zv[3], h1[64], h2[64], rs[3];
    if (t < 25) ns[t] = nsrc[b * 25 + t];
    if (t < 3)  zv[t] = z[b * 3 + t];
    __syncthreads();

    // h1 = elu(W1 z + B1)
    float a1 = gen_g(192 + t, ns, w1W, w1b, b1W, b1b, w2W, w2b, b2W, b2b,
                     w3W, w3b, b3W, b3b);
    #pragma unroll
    for (int c = 0; c < 3; ++c)
        a1 += gen_g(t * 3 + c, ns, w1W, w1b, b1W, b1b, w2W, w2b, b2W, b2b,
                    w3W, w3b, b3W, b3b) * zv[c];
    h1[t] = eluf(a1);
    __syncthreads();

    // h2 = elu(W2 h1 + B2)
    float a2 = gen_g(4352 + t, ns, w1W, w1b, b1W, b1b, w2W, w2b, b2W, b2b,
                     w3W, w3b, b3W, b3b);
    for (int c = 0; c < 64; ++c)
        a2 += gen_g(256 + t * 64 + c, ns, w1W, w1b, b1W, b1b, w2W, w2b,
                    b2W, b2b, w3W, w3b, b3W, b3b) * h1[c];
    h2[t] = eluf(a2);
    __syncthreads();

    // z_hat = W3 h2 + B3
    if (t < 3) {
        float a3 = gen_g(4608 + t, ns, w1W, w1b, b1W, b1b, w2W, w2b, b2W, b2b,
                         w3W, w3b, b3W, b3b);
        for (int c = 0; c < 64; ++c)
            a3 += gen_g(4416 + t * 64 + c, ns, w1W, w1b, b1W, b1b, w2W, w2b,
                        b2W, b2b, w3W, w3b, b3W, b3b) * h2[c];
        out[2 + b * 3 + t] = a3;
        float dz = a3 - z_next[b * 3 + t];
        rs[t] = dz * dz;
    }
    __syncthreads();
    if (t == 0) atomicAdd(out, (rs[0] + rs[1] + rs[2]) * (1.0f / (float)BATCH));
}

// ---------------------------------------------------------------------------
// Pairwise-distance GEMM with fused column-min (the heavy kernel).
//   S[i,j] = A_i . B_j  (bf16 inputs, fp32 WMMA accumulate)
//   colmin[j] = min over i of (anorm[i] - 2*S[i,j])   [order-uint keys]
// Block: 128x128 tile, 256 threads = 8 waves arranged 4(M) x 2(N);
// each wave: 32x64 tile = 2 A-frags x 4 B-frags -> 8 wmma per 32-K step.
// Double-buffered async global->LDS staging (ASYNCcnt pipeline).
// ---------------------------------------------------------------------------
__device__ __forceinline__ v16bf load_frag(const unsigned short* row_base, int h) {
    union { uint4 u[2]; v16bf v; } f;
    f.u[0] = *(const uint4*)(row_base + 8 * h);        // K = 8h .. 8h+7
    f.u[1] = *(const uint4*)(row_base + 16 + 8 * h);   // K = 16+8h .. 16+8h+7
    return f.v;
}

#define LDSROW 40   // bf16 row stride: 80B -> conflict-free b128 fragment loads

__global__ __launch_bounds__(256) void pairmin_kernel(
    const unsigned short* __restrict__ Abf,   // [8192][4640] bf16
    const unsigned short* __restrict__ Bbf,   // [8192][4640] bf16
    const float* __restrict__ anorm,          // [8192] ||A_i||^2
    unsigned* __restrict__ colmin,            // [8192] ordered-uint min keys
    int excludeDiag)
{
    const int tid  = threadIdx.x;
    const int wid  = tid >> 5;
    const int lane = tid & 31;
    const int h    = lane >> 4;
    const int m    = lane & 15;
    const int wm   = wid & 3;        // wave row block (32 rows)
    const int wn   = wid >> 2;       // wave col block (64 cols)
    const int i0   = blockIdx.y * 128;
    const int j0   = blockIdx.x * 128;

    __shared__ unsigned short As[2][128 * LDSROW];
    __shared__ unsigned short Bs[2][128 * LDSROW];
    __shared__ float    anS[128];
    __shared__ unsigned cmin[128];

    if (tid < 128) { anS[tid] = anorm[i0 + tid]; cmin[tid] = 0xFFFFFFFFu; }

    // staging: 512 x 16B chunks per tile; thread t covers chunks t, t+256
    // (4 async instructions per wave per K-step => wait thresholds 4 / 0)
    const int row0 = tid >> 2,         ch0 = (tid & 3) * 8;
    const int row1 = (tid + 256) >> 2, ch1 = ((tid + 256) & 3) * 8;

    #define STAGE(buf, ks)                                                     \
        do {                                                                   \
            const int kb_ = (ks) * 32;                                         \
            cp16_to_lds(Abf + (size_t)(i0 + row0) * KPAD + kb_ + ch0,          \
                        &As[buf][row0 * LDSROW + ch0]);                        \
            cp16_to_lds(Bbf + (size_t)(j0 + row0) * KPAD + kb_ + ch0,          \
                        &Bs[buf][row0 * LDSROW + ch0]);                        \
            cp16_to_lds(Abf + (size_t)(i0 + row1) * KPAD + kb_ + ch1,          \
                        &As[buf][row1 * LDSROW + ch1]);                        \
            cp16_to_lds(Bbf + (size_t)(j0 + row1) * KPAD + kb_ + ch1,          \
                        &Bs[buf][row1 * LDSROW + ch1]);                        \
        } while (0)

    v8f acc[8];                       // acc[mi*4 + jt]
    #pragma unroll
    for (int t = 0; t < 8; ++t) acc[t] = (v8f){};

    STAGE(0, 0);
    STAGE(1, 1);

    for (int ks = 0; ks < KSTEPS; ++ks) {
        const int cur = ks & 1;
        // drain this tile's async loads (next tile may stay in flight)
        wait_async_le(ks + 1 < KSTEPS ? 4 : 0);
        __syncthreads();               // tile ks visible to all waves

        // pre-load all 6 fragments, then issue the 8-wmma burst
        v16bf af0 = load_frag(&As[cur][(wm * 32 +  0 + m) * LDSROW], h);
        v16bf af1 = load_frag(&As[cur][(wm * 32 + 16 + m) * LDSROW], h);
        v16bf bf0 = load_frag(&Bs[cur][(wn * 64 +  0 + m) * LDSROW], h);
        v16bf bf1 = load_frag(&Bs[cur][(wn * 64 + 16 + m) * LDSROW], h);
        v16bf bf2 = load_frag(&Bs[cur][(wn * 64 + 32 + m) * LDSROW], h);
        v16bf bf3 = load_frag(&Bs[cur][(wn * 64 + 48 + m) * LDSROW], h);

        acc[0] = __builtin_amdgcn_wmma_f32_16x16x32_bf16(false, af0, false, bf0, (short)0, acc[0], false, false);
        acc[1] = __builtin_amdgcn_wmma_f32_16x16x32_bf16(false, af0, false, bf1, (short)0, acc[1], false, false);
        acc[2] = __builtin_amdgcn_wmma_f32_16x16x32_bf16(false, af0, false, bf2, (short)0, acc[2], false, false);
        acc[3] = __builtin_amdgcn_wmma_f32_16x16x32_bf16(false, af0, false, bf3, (short)0, acc[3], false, false);
        acc[4] = __builtin_amdgcn_wmma_f32_16x16x32_bf16(false, af1, false, bf0, (short)0, acc[4], false, false);
        acc[5] = __builtin_amdgcn_wmma_f32_16x16x32_bf16(false, af1, false, bf1, (short)0, acc[5], false, false);
        acc[6] = __builtin_amdgcn_wmma_f32_16x16x32_bf16(false, af1, false, bf2, (short)0, acc[6], false, false);
        acc[7] = __builtin_amdgcn_wmma_f32_16x16x32_bf16(false, af1, false, bf3, (short)0, acc[7], false, false);

        __syncthreads();               // all waves done reading buffer `cur`
        if (ks + 2 < KSTEPS) STAGE(cur, ks + 2);
    }
    __syncthreads();

    // fused epilogue: per-column min of (anorm[i] - 2*S[i,j])
    #pragma unroll
    for (int mi = 0; mi < 2; ++mi) {
        #pragma unroll
        for (int jt = 0; jt < 4; ++jt) {
            const int j = j0 + wn * 64 + jt * 16 + m;
            float lmin = 3.4e38f;
            #pragma unroll
            for (int r = 0; r < 8; ++r) {
                int ir = wm * 32 + mi * 16 + 8 * h + r;  // C layout: M = r + 8h
                float key = anS[ir] - 2.0f * acc[mi * 4 + jt][r];
                if (excludeDiag && (i0 + ir) == j) key = 3.4e38f;
                lmin = fminf(lmin, key);
            }
            lmin = fminf(lmin, __shfl_xor(lmin, 16));  // other row half, same col
            if (h == 0) atomicMin(&cmin[wn * 64 + jt * 16 + m], f2ord(lmin));
        }
    }
    __syncthreads();
    if (tid < 128) atomicMin(&colmin[j0 + tid], cmin[tid]);
    #undef STAGE
}

// ---------------------------------------------------------------------------
// KL finalize: wp/ww from min keys + own norm, sqrt, log, mean * D
// ---------------------------------------------------------------------------
__global__ __launch_bounds__(256) void kl_kernel(
    const unsigned* __restrict__ wpm, const unsigned* __restrict__ wwm,
    const float* __restrict__ gn, float* __restrict__ out)
{
    int j = blockIdx.x * 256 + threadIdx.x;
    float term = 0.f;
    if (j < BATCH) {
        float gnj = gn[j];
        float wp = sqrtf(fmaxf(ord2f(wpm[j]) + gnj, 0.f) + 1e-8f);
        float ww = sqrtf(fmaxf(ord2f(wwm[j]) + gnj, 0.f) + 1e-8f);
        term = logf(wp / (ww + 1e-8f) + 1e-8f);
    }
    float s = wave_reduce_sum(term);
    if ((threadIdx.x & 31) == 0)
        atomicAdd(out + 1, s * ((float)DFEAT / (float)BATCH));
}

// ---------------------------------------------------------------------------
// Entry point
// ---------------------------------------------------------------------------
extern "C" void kernel_launch(void* const* d_in, const int* in_sizes, int n_in,
                              void* d_out, int out_size, void* d_ws, size_t ws_size,
                              hipStream_t stream) {
    const float* z      = (const float*)d_in[0];
    const float* z_next = (const float*)d_in[1];
    const float* n      = (const float*)d_in[2];
    const float* prior  = (const float*)d_in[3];
    const float* w1W = (const float*)d_in[4];  const float* w1b = (const float*)d_in[5];
    const float* b1W = (const float*)d_in[6];  const float* b1b = (const float*)d_in[7];
    const float* w2W = (const float*)d_in[8];  const float* w2b = (const float*)d_in[9];
    const float* b2W = (const float*)d_in[10]; const float* b2b = (const float*)d_in[11];
    const float* w3W = (const float*)d_in[12]; const float* w3b = (const float*)d_in[13];
    const float* b3W = (const float*)d_in[14]; const float* b3b = (const float*)d_in[15];
    float* out = (float*)d_out;

    char* ws = (char*)d_ws;
    const size_t szMat = (size_t)BATCH * KPAD * 2;     // 76,021,760 B each
    unsigned short* Gb = (unsigned short*)ws;
    unsigned short* Pb = (unsigned short*)(ws + szMat);
    float*    gn  = (float*)(ws + 2 * szMat);
    float*    pn  = gn + BATCH;
    unsigned* wpm = (unsigned*)(pn + BATCH);
    unsigned* wwm = wpm + BATCH;

    init_kernel<<<32, 256, 0, stream>>>(gn, pn, wpm, wwm, out);

    hyper_kernel<<<dim3(19, BATCH), 256, 0, stream>>>(
        n, w1W, w1b, b1W, b1b, w2W, w2b, b2W, b2b, w3W, w3b, b3W, b3b, Gb, gn);

    prior_kernel<<<dim3(19, BATCH), 256, 0, stream>>>(prior, Pb, pn);

    decode_kernel<<<BATCH, 64, 0, stream>>>(
        z, z_next, n, w1W, w1b, b1W, b1b, w2W, w2b, b2W, b2b, w3W, w3b, b3W, b3b, out);

    // wp: min over prior rows i of dist(prior_i, G_j)
    pairmin_kernel<<<dim3(64, 64), 256, 0, stream>>>(Pb, Gb, pn, wpm, 0);
    // ww: min over G rows i != j of dist(G_i, G_j)
    pairmin_kernel<<<dim3(64, 64), 256, 0, stream>>>(Gb, Gb, gn, wwm, 1);

    kl_kernel<<<32, 256, 0, stream>>>(wpm, wwm, gn, out);
}
